// Hippocampus_43808666419586
// MI455X (gfx1250) — matrix operand order repaired and checked by reading, not compiled, not run on hardware
//
#include <hip/hip_runtime.h>
#include <math.h>

// ---------------------------------------------------------------------------
// MI455X (gfx1250) implementation.
//
// Roofline: 262144 queries * ~155 KFLOP = ~40.6 GFLOP vs ~236 MB of HBM
// traffic (~10 us @ 23.3 TB/s)  ->  compute bound. The two MLP layers
// (160->256->128) and the 16-slot cosine match are batched GEMMs; we run
// them on v_wmma_f32_16x16x32_f16 (f16 operands, f32 accumulate).
// Per-query data is streamed into LDS with GLOBAL_LOAD_ASYNC_TO_LDS_B128
// (ASYNCcnt) when the toolchain exposes the builtin; everything
// data-dependent (episodic gather, gate MLP, 32x44 reinstatement GEMV)
// stays on VALU in an epilog (~5 KFLOP/query).
// ---------------------------------------------------------------------------

typedef __attribute__((ext_vector_type(16))) _Float16 v16h;
typedef __attribute__((ext_vector_type(8)))  _Float16 h8;
typedef __attribute__((ext_vector_type(8)))  float    v8f;
typedef __attribute__((ext_vector_type(4)))  float    v4f;
typedef _Float16 half_t;

// Matches the builtin's parameter type exactly (per hipcc diagnostic):
//   'int __attribute__((vector_size(16))) __device__ *'  (addrspace(1))
//   and the LDS side is the addrspace(3) flavor.
typedef int v4i __attribute__((vector_size(16)));
typedef __attribute__((address_space(1))) v4i* gptr_v4i;
typedef __attribute__((address_space(3))) v4i* lptr_v4i;

// Probe the gfx1250 async global->LDS builtins (declared for gfx1250 in this
// toolchain; fall back to a plain vector copy otherwise).
#if defined(__has_builtin)
#  if __has_builtin(__builtin_amdgcn_global_load_async_to_lds_b128)
#    define HIPPO_ASYNC 1
#  endif
#endif
#ifndef HIPPO_ASYNC
#  define HIPPO_ASYNC 0
#endif

// LDS layout (bytes). Dynamic shared memory; row strides padded so that the
// 16 rows of a wave-tile land on distinct bank groups and stay 16B aligned.
#define COMB_STRIDE 168   // 160 cols + pad ; 336 B rows (21*16)
#define H_STRIDE    264   // 256 cols + pad ; 528 B rows (33*16)
#define KEY_STRIDE  136   // 128 cols + pad ; 272 B rows (17*16)
#define SIMS_STRIDE 20    // 16 cols + pad  ; 80 B rows

#define COMB_OFF 0                         // 128*168*2      = 43008 B (dead after layer1)
#define SIMS_OFF 0                         // 8*16*20*4      = 10240 B (overlays comb)
#define INV_OFF  10240                     // 8*16*4         =   512 B (overlays comb)
#define H_OFF    43008                     // 8*16*264*2     = 67584 B
#define KEY_OFF  (43008 + 67584)           // 8*16*136*2     = 34816 B
#define LDS_BYTES (43008 + 67584 + 34816)  // 145408 B  (2 blocks fit in 320KB WGP LDS)

// Transient f32 staging buffer for async copies: 128 rows * 160 f32 = 81920 B.
// Overlays the h+key regions (102400 B), which are only written after the
// staging barriers retire all reads of it.
#define COMB32_OFF H_OFF

// Weight-fragment sizes in d_ws (f16 halves), pre-swizzled to B-matrix lane
// order: frag[(tile*32 + lane)*16 + e] with e = contiguous K within lane.
#define W1F_HALVES (16*5*512)   // 16 N-tiles x 5 K-tiles (160x256)
#define W2F_HALVES (8*8*512)    //  8 N-tiles x 8 K-tiles (256x128)
#define PFR_HALVES (4*512)      //  4 K-tiles            (128x16, pre-normalized)
#define PREP_TOTAL (W1F_HALVES + W2F_HALVES + PFR_HALVES)

// ---------------------------------------------------------------------------
// Prep kernel: cast + swizzle weights into WMMA B-fragment order, and
// L2-normalize the prototype rows. Deterministic, re-run every launch.
// B-fragment k order used by this code (A loads match it): lane = column n (n = lane%16),
// g = lane>>4 selects K half; VGPR r half p hold k_local = 16*g + 2r + p,
// i.e. each lane's 16 fragment halves are 16 *consecutive* K values.
// ---------------------------------------------------------------------------
__global__ __launch_bounds__(256) void hippo_prep(
    const float* __restrict__ W1, const float* __restrict__ W2,
    const float* __restrict__ protos,
    half_t* __restrict__ w1f, half_t* __restrict__ w2f, half_t* __restrict__ pfr)
{
  int idx = blockIdx.x * blockDim.x + threadIdx.x;
  if (idx < W1F_HALVES) {
    int nt   = idx / (5*512);
    int rem  = idx % (5*512);
    int kt   = rem / 512;
    int lid  = rem % 512;
    int lane = lid >> 4, e = lid & 15, g = lane >> 4;
    int n = nt*16 + (lane & 15);
    int k = kt*32 + 16*g + e;               // k < 160
    w1f[idx] = (half_t)W1[n*160 + k];       // W1 is (256,160): B[k][n] = W1[n][k]
  } else if (idx < W1F_HALVES + W2F_HALVES) {
    int j    = idx - W1F_HALVES;
    int nt   = j / (8*512);
    int rem  = j % (8*512);
    int kt   = rem / 512;
    int lid  = rem % 512;
    int lane = lid >> 4, e = lid & 15, g = lane >> 4;
    int n = nt*16 + (lane & 15);
    int k = kt*32 + 16*g + e;               // k < 256
    w2f[j] = (half_t)W2[n*256 + k];         // W2 is (128,256)
  } else if (idx < PREP_TOTAL) {
    int j    = idx - (W1F_HALVES + W2F_HALVES);
    int kt   = j / 512;
    int lid  = j % 512;
    int lane = lid >> 4, e = lid & 15, g = lane >> 4;
    int n = lane & 15;                      // 16 slots = exactly one N-tile
    int k = kt*32 + 16*g + e;               // k < 128
    float ss = 0.0f;
    for (int c = 0; c < 128; ++c) { float v = protos[n*128 + c]; ss += v*v; }
    float inv = 1.0f / fmaxf(sqrtf(ss), 1e-12f);
    pfr[j] = (half_t)(protos[n*128 + k] * inv);
  }
}

// A-fragment (16x32 f16) loader from a row-major f16 LDS tile.
// Layout: lane holds row m = lane%16; VGPRs 0-3 hold K = 16g..16g+7,
// VGPRs 4-7 hold K = 16g+8..16g+15 (g = lane>>4), matching B. Both chunks are 16B
// aligned given our padded strides -> two ds_load_b128.
__device__ __forceinline__ v16h load_a_frag(const half_t* row_kt_base, int g) {
  union { v16h v; h8 h[2]; } u;
  u.h[0] = *(const h8*)(row_kt_base + 16*g);
  u.h[1] = *(const h8*)(row_kt_base + 16*g + 8);
  return u.v;
}

// B-fragment loader from pre-swizzled global fragments (16 contiguous halves
// per lane -> two global b128 loads; same address across all waves/blocks,
// resident in L2).
__device__ __forceinline__ v16h load_b_frag(const half_t* p) {
  union { v16h v; h8 h[2]; } u;
  u.h[0] = *(const h8*)(p);
  u.h[1] = *(const h8*)(p + 8);
  return u.v;
}

// ---------------------------------------------------------------------------
// Main kernel: 256 threads = 8 wave32, 128 queries per block.
// Wave w owns query rows [w*16, w*16+16).
// ---------------------------------------------------------------------------
__global__ __launch_bounds__(256) void hippo_main(
    const float* __restrict__ act,  const float* __restrict__ pfc,
    const float* __restrict__ ctd,  const float* __restrict__ logT,
    const float* __restrict__ b1,   const float* __restrict__ b2,
    const float* __restrict__ episodes, const float* __restrict__ tderr,
    const float* __restrict__ gW1,  const float* __restrict__ gb1,
    const float* __restrict__ gW2,  const float* __restrict__ gb2,
    const float* __restrict__ rpW,  const float* __restrict__ rpb,
    const float* __restrict__ rnW,  const float* __restrict__ rnb,
    const half_t* __restrict__ w1f, const half_t* __restrict__ w2f,
    const half_t* __restrict__ pfr,
    float* __restrict__ out)
{
  extern __shared__ char smem[];
  half_t* comb   = (half_t*)(smem + COMB_OFF);
  half_t* hbuf   = (half_t*)(smem + H_OFF);
  half_t* kbuf   = (half_t*)(smem + KEY_OFF);
  float*  sims   = (float*)(smem + SIMS_OFF);
  float*  invn   = (float*)(smem + INV_OFF);
  float*  comb32 = (float*)(smem + COMB32_OFF);

  const int tid  = threadIdx.x;
  const int lane = tid & 31;
  const int w    = tid >> 5;           // wave id 0..7
  const int q0   = blockIdx.x * 128;   // first query of this block

  // ---- Phase -1: async-stage raw f32 [activation(128)|pfc(32)] rows -----
  // 128 rows * 40 16-byte chunks = 5120 async b128 transfers per block.
  for (int i = tid; i < 128 * 40; i += 256) {
    int q = i / 40, j = i - q * 40;            // j = 16B chunk within row
    const float* src = (j < 32)
        ? act + ((size_t)(q0 + q) * 128 + j * 4)
        : pfc + ((size_t)(q0 + q) * 32 + (j - 32) * 4);
    float* dst = comb32 + q * 160 + j * 4;
#if HIPPO_ASYNC
    __builtin_amdgcn_global_load_async_to_lds_b128(
        (gptr_v4i)const_cast<float*>(src),   // addrspace(1) int4*
        (lptr_v4i)dst,                       // addrspace(3) int4*
        0, 0);
#else
    *(v4f*)dst = *(const v4f*)src;
#endif
  }
#if HIPPO_ASYNC
#  if __has_builtin(__builtin_amdgcn_s_wait_asynccnt)
  __builtin_amdgcn_s_wait_asynccnt(0);
#  else
  asm volatile("s_wait_asynccnt 0" ::: "memory");
#  endif
#endif
  __syncthreads();

  // ---- Phase 0: convert staged rows to f16 into the padded comb tile ----
  for (int i = tid; i < 128 * 160; i += 256) {
    int q = i / 160, c = i - q * 160;
    comb[q * COMB_STRIDE + c] = (half_t)comb32[q * 160 + c];
  }
  __syncthreads();   // conversion reads done before hbuf/kbuf overwrite comb32

  const int m = lane & 15;   // tile row (A) / column n (B,C,D)
  const int g = lane >> 4;

  // ---- Phase 1: h = relu(comb @ W1^T + b1) -> f16 LDS -------------------
  for (int nt = 0; nt < 16; ++nt) {
    v8f acc = {};
    const float bias = b1[nt * 16 + m];
    #pragma unroll
    for (int kt = 0; kt < 5; ++kt) {
      v16h a  = load_a_frag(comb + (w * 16 + m) * COMB_STRIDE + kt * 32, g);
      v16h bm = load_b_frag(w1f + ((nt * 5 + kt) * 32 + lane) * 16);
      acc = __builtin_amdgcn_wmma_f32_16x16x32_f16(false, a, false, bm,
                                                   (short)0, acc, false, false);
    }
    // D(r,lane) = element (row r+8g, col nt*16+m); relu + scatter to LDS.
    #pragma unroll
    for (int r = 0; r < 8; ++r) {
      float v = acc[r] + bias;
      v = v > 0.0f ? v : 0.0f;
      hbuf[(w * 16 + r + 8 * g) * H_STRIDE + nt * 16 + m] = (half_t)v;
    }
  }
  __syncthreads();   // also retires comb reads before sims/invn overlay it

  // ---- Phase 2: key = h @ W2^T + b2 -> f16 LDS --------------------------
  for (int nt = 0; nt < 8; ++nt) {
    v8f acc = {};
    const float bias = b2[nt * 16 + m];
    #pragma unroll
    for (int kt = 0; kt < 8; ++kt) {
      v16h a  = load_a_frag(hbuf + (w * 16 + m) * H_STRIDE + kt * 32, g);
      v16h bm = load_b_frag(w2f + ((nt * 8 + kt) * 32 + lane) * 16);
      acc = __builtin_amdgcn_wmma_f32_16x16x32_f16(false, a, false, bm,
                                                   (short)0, acc, false, false);
    }
    #pragma unroll
    for (int r = 0; r < 8; ++r)
      kbuf[(w * 16 + r + 8 * g) * KEY_STRIDE + nt * 16 + m] =
          (half_t)(acc[r] + bias);
  }

  // ---- Phase 3a: inverse row norms of key (lanes 0..15, own-wave tile) --
  if (lane < 16) {
    float ss = 0.0f;
    const half_t* row = kbuf + (w * 16 + lane) * KEY_STRIDE;
    #pragma unroll
    for (int c = 0; c < 128; c += 8) {
      h8 hv = *(const h8*)(row + c);
      #pragma unroll
      for (int j = 0; j < 8; ++j) { float f = (float)hv[j]; ss += f * f; }
    }
    invn[w * 16 + lane] = 1.0f / fmaxf(sqrtf(ss), 1e-12f);
  }

  // ---- Phase 3b: sims = normalize(key) @ protoN^T (WMMA + row scale) ----
  {
    v8f acc = {};
    #pragma unroll
    for (int kt = 0; kt < 4; ++kt) {
      v16h a  = load_a_frag(kbuf + (w * 16 + m) * KEY_STRIDE + kt * 32, g);
      v16h bm = load_b_frag(pfr + (kt * 32 + lane) * 16);
      acc = __builtin_amdgcn_wmma_f32_16x16x32_f16(false, a, false, bm,
                                                   (short)0, acc, false, false);
    }
    #pragma unroll
    for (int r = 0; r < 8; ++r) {
      int row = w * 16 + r + 8 * g;
      sims[row * SIMS_STRIDE + m] = acc[r] * invn[row];
    }
  }
  __syncthreads();

  // ---- Phase 4: per-query epilog (thread t handles query t, t < 128) ----
  if (tid < 128) {
    const int q = tid;
    const size_t gq = (size_t)q0 + q;
    const float* srow = sims + q * SIMS_STRIDE;

    float s[16];
    #pragma unroll
    for (int i = 0; i < 16; ++i) s[i] = srow[i];

    int slot = 0; float bestv = s[0];
    #pragma unroll
    for (int i = 1; i < 16; ++i)
      if (s[i] > bestv) { bestv = s[i]; slot = i; }   // first-max, matches argmax

    // softmax(sims * exp(logT))
    const float T = __expf(logT[0]);
    float soft[16]; float ssum = 0.0f;
    #pragma unroll
    for (int i = 0; i < 16; ++i) {
      float e = __expf((s[i] - bestv) * T);
      soft[i] = e; ssum += e;
    }
    const float rs = 1.0f / ssum;

    // pfc row (f32) + inverse norm
    float pv[32]; float pss = 0.0f;
    #pragma unroll
    for (int j = 0; j < 32; ++j) { pv[j] = pfc[gq * 32 + j]; pss += pv[j] * pv[j]; }
    const float pinv = 1.0f / fmaxf(sqrtf(pss), 1e-12f);

    // episodic retrieval for the chosen slot
    const float* eb    = episodes + slot * (8 * 44);
    const float* tdrow = tderr + slot * 8;
    int beste = 0; float bestr = -1e30f;
    for (int e = 0; e < 8; ++e) {
      const float* ep = eb + e * 44;
      float dot = 0.0f, ess = 0.0f;
      #pragma unroll
      for (int j = 0; j < 32; ++j) { float ev = ep[j]; dot += ev * pv[j]; ess += ev * ev; }
      float einv = 1.0f / fmaxf(sqrtf(ess), 1e-12f);
      float rel = dot * einv * pinv * fmaxf(fabsf(tdrow[e]), 1e-6f);
      if (rel > bestr) { bestr = rel; beste = e; }
    }
    const float* epbest = eb + beste * 44;
    float ec[44];
    #pragma unroll
    for (int j = 0; j < 44; ++j) ec[j] = epbest[j];
    const float ep_td = tdrow[beste];

    // read gate MLP: 3 -> 16 -> 1, tanh/tanh
    const float gi0 = bestv, gi1 = fabsf(ctd[gq]), gi2 = ep_td;
    float gacc = gb2[0];
    #pragma unroll
    for (int u = 0; u < 16; ++u) {
      float hu = tanhf(gW1[u * 3 + 0] * gi0 + gW1[u * 3 + 1] * gi1 +
                       gW1[u * 3 + 2] * gi2 + gb1[u]);
      gacc += gW2[u] * hu;
    }
    const float alpha = tanhf(gacc);

    float* o = out + gq * 61;
    o[0] = alpha;

    // new_pfc = pfc + alpha * (rp_W @ ep + rp_b)
    for (int j = 0; j < 32; ++j) {
      float d = rpb[j];
      #pragma unroll
      for (int c2 = 0; c2 < 44; ++c2) d += rpW[j * 44 + c2] * ec[c2];
      o[1 + j] = pv[j] + alpha * d;
    }

    // neuromodulators: nm[grp*4+p]; eta/decay clip [0.1,1], expl [0.1,0.5]
    for (int i = 0; i < 12; ++i) {
      float d = rnb[i];
      #pragma unroll
      for (int c2 = 0; c2 < 44; ++c2) d += rnW[i * 44 + c2] * ec[c2];
      float hi = (i >= 8) ? 0.5f : 1.0f;
      o[33 + i] = fminf(fmaxf(d, 0.1f), hi);
    }

    // straight-through one-hot: hard - soft + soft (forward arithmetic kept)
    #pragma unroll
    for (int i = 0; i < 16; ++i) {
      float hard = (i == slot) ? 1.0f : 0.0f;
      float sv = soft[i] * rs;
      o[45 + i] = (hard - sv) + sv;
    }
  }
}

extern "C" void kernel_launch(void* const* d_in, const int* in_sizes, int n_in,
                              void* d_out, int out_size, void* d_ws, size_t ws_size,
                              hipStream_t stream) {
  const float* act    = (const float*)d_in[0];
  const float* pfc    = (const float*)d_in[1];
  const float* ctd    = (const float*)d_in[2];
  const float* protos = (const float*)d_in[3];
  const float* logT   = (const float*)d_in[4];
  const float* W1     = (const float*)d_in[5];
  const float* b1     = (const float*)d_in[6];
  const float* W2     = (const float*)d_in[7];
  const float* b2     = (const float*)d_in[8];
  const float* eps    = (const float*)d_in[9];
  const float* tderr  = (const float*)d_in[10];
  const float* gW1    = (const float*)d_in[11];
  const float* gb1    = (const float*)d_in[12];
  const float* gW2    = (const float*)d_in[13];
  const float* gb2    = (const float*)d_in[14];
  const float* rpW    = (const float*)d_in[15];
  const float* rpb    = (const float*)d_in[16];
  const float* rnW    = (const float*)d_in[17];
  const float* rnb    = (const float*)d_in[18];

  half_t* w1f = (half_t*)d_ws;
  half_t* w2f = w1f + W1F_HALVES;
  half_t* pfr = w2f + W2F_HALVES;

  hippo_prep<<<(PREP_TOTAL + 255) / 256, 256, 0, stream>>>(W1, W2, protos,
                                                           w1f, w2f, pfr);

  const int nB = in_sizes[2];          // current_td has B elements
  const int nblocks = nB / 128;

  // 142 KB dynamic LDS per block (2 blocks per 320KB WGP).
  (void)hipFuncSetAttribute((const void*)hippo_main,
                            hipFuncAttributeMaxDynamicSharedMemorySize,
                            LDS_BYTES);

  hippo_main<<<nblocks, 256, LDS_BYTES, stream>>>(
      act, pfc, ctd, logT, b1, b2, eps, tderr, gW1, gb1, gW2, gb2,
      rpW, rpb, rnW, rnb, w1f, w2f, pfr, (float*)d_out);
}
